// PQBASTLinear_82291573391728
// MI455X (gfx1250) — matrix operand order repaired and checked by reading, not compiled, not run on hardware
//
#include <hip/hip_runtime.h>

// ---------------------------------------------------------------------------
// Problem constants (from the reference)
// ---------------------------------------------------------------------------
#define BB   16
#define SS   4096
#define DIN  1024
#define DOUT 1024
#define RR   16
#define TR   16
#define NA   4
#define SCALING 1.0f

typedef __bf16  v16bf __attribute__((ext_vector_type(16)));
typedef __bf16  v8bf  __attribute__((ext_vector_type(8)));
typedef float   v8f   __attribute__((ext_vector_type(8)));
typedef int     v4i   __attribute__((ext_vector_type(4)));

// CDNA5 async global->LDS path (ASYNCcnt), with portable fallback
#if __has_builtin(__builtin_amdgcn_global_load_async_to_lds_b128) && \
    __has_builtin(__builtin_amdgcn_s_wait_asynccnt)
#define USE_ASYNC_LDS 1
#else
#define USE_ASYNC_LDS 0
#endif

// ---------------------------------------------------------------------------
// Step 1a: t1[n] = W_a[n] @ W_s  (16x16),  t2[n] = W_q @ W_b[n] (16x16)
// ---------------------------------------------------------------------------
__global__ void small_mm1(const float* __restrict__ W_a,   // [NA,16,DIN]
                          const float* __restrict__ W_s,   // [DIN,16]
                          const float* __restrict__ W_q,   // [16,DOUT]
                          const float* __restrict__ W_b,   // [NA,DOUT,16]
                          float* __restrict__ t1, float* __restrict__ t2) {
  const int n = blockIdx.x;
  const int t = threadIdx.x;
  const int r = t >> 4, c = t & 15;
  const float* wa = W_a + ((size_t)n * 16 + r) * DIN;
  const float* wq = W_q + (size_t)r * DOUT;
  const float* wb = W_b + (size_t)n * DOUT * 16 + c;
  float s1 = 0.f, s2 = 0.f;
  for (int k = 0; k < DIN; ++k) {
    s1 += wa[k] * W_s[k * 16 + c];
    s2 += wq[k] * wb[(size_t)k * 16];
  }
  t1[(n * 16 + r) * 16 + c] = s1;
  t2[(n * 16 + r) * 16 + c] = s2;
}

// ---------------------------------------------------------------------------
// Step 1b: Ap[n] = t1[n] @ W_t  (16x1024),  Bp[n] = W_p @ t2[n]  (1024x16)
// ---------------------------------------------------------------------------
__global__ void small_mm2(const float* __restrict__ W_t,   // [16,DIN]
                          const float* __restrict__ W_p,   // [DOUT,16]
                          const float* __restrict__ t1, const float* __restrict__ t2,
                          float* __restrict__ Ap, float* __restrict__ Bp) {
  const int idx = blockIdx.x * 256 + threadIdx.x;    // 131072 threads
  if (idx < NA * 16 * DIN) {                         // Ap half
    const int n = idx >> 14, r = (idx >> 10) & 15, i = idx & (DIN - 1);
    const float* t1n = t1 + (n * 16 + r) * 16;
    float s = 0.f;
#pragma unroll
    for (int j = 0; j < 16; ++j) s += t1n[j] * W_t[j * DIN + i];
    Ap[idx] = s;                                     // Ap[n][r][i]
  } else {                                           // Bp half
    const int id2 = idx - NA * 16 * DIN;
    const int n = id2 >> 14, o = (id2 >> 4) & (DOUT - 1), c = id2 & 15;
    const float* wp  = W_p + (size_t)o * 16;
    const float* t2n = t2 + n * 256;
    float s = 0.f;
#pragma unroll
    for (int j = 0; j < 16; ++j) s += wp[j] * t2n[j * 16 + c];
    Bp[id2] = s;                                     // Bp[n][o][c]
  }
}

// ---------------------------------------------------------------------------
// Step 2: Wc[n][o][i] = bf16( W_base[o][i] + SCALING * sum_r Bp[n][o][r]*Ap[n][r][i] )
// ---------------------------------------------------------------------------
__global__ void build_wc(const float* __restrict__ W_base,
                         const float* __restrict__ Ap, const float* __restrict__ Bp,
                         __bf16* __restrict__ Wc) {
  const int idx = blockIdx.x * 256 + threadIdx.x;    // 4M threads
  const int n = idx >> 20;
  const int rem = idx & ((1 << 20) - 1);
  const int o = rem >> 10, i = rem & (DIN - 1);
  const float* bp = Bp + ((size_t)n * DOUT + o) * 16;
  const float* ap = Ap + (size_t)n * 16 * DIN + i;
  float s = 0.f;
#pragma unroll
  for (int r = 0; r < 16; ++r) s += bp[r] * ap[(size_t)r * DIN];
  Wc[idx] = (__bf16)(W_base[rem] + SCALING * s);
}

// ---------------------------------------------------------------------------
// Step 3: main GEMM, double-buffered LDS, async W-tile copy, bf16 WMMA
// ---------------------------------------------------------------------------
#define KSTEP 32
#define LDS_STRIDE 40   // 32 + 8 bf16 pad -> 80B rows, 16B aligned, conflict-free

static __device__ __forceinline__
void stage_x(__bf16 (*bx)[LDS_STRIDE], const float* __restrict__ xrow,
             int k0, int srow, int shalf) {
  float4 f0 = *(const float4*)(xrow + k0);
  float4 f1 = *(const float4*)(xrow + k0 + 4);
  float4 f2 = *(const float4*)(xrow + k0 + 8);
  float4 f3 = *(const float4*)(xrow + k0 + 12);
  v8bf xa, xb;
  xa[0]=(__bf16)f0.x; xa[1]=(__bf16)f0.y; xa[2]=(__bf16)f0.z; xa[3]=(__bf16)f0.w;
  xa[4]=(__bf16)f1.x; xa[5]=(__bf16)f1.y; xa[6]=(__bf16)f1.z; xa[7]=(__bf16)f1.w;
  xb[0]=(__bf16)f2.x; xb[1]=(__bf16)f2.y; xb[2]=(__bf16)f2.z; xb[3]=(__bf16)f2.w;
  xb[4]=(__bf16)f3.x; xb[5]=(__bf16)f3.y; xb[6]=(__bf16)f3.z; xb[7]=(__bf16)f3.w;
  *(v8bf*)&bx[srow][shalf]     = xa;
  *(v8bf*)&bx[srow][shalf + 8] = xb;
}

static __device__ __forceinline__
void stage_w(__bf16 (*bw)[LDS_STRIDE], const __bf16* __restrict__ wrow,
             int k0, int srow, int shalf) {
#if USE_ASYNC_LDS
  // DMA 2x16B directly into LDS; tracked by ASYNCcnt (no VGPR round trip).
  // Builtin signature (from compiler diagnostic): (v4i as1*, v4i as3*, imm, imm)
  __builtin_amdgcn_global_load_async_to_lds_b128(
      (__attribute__((address_space(1))) v4i*)(wrow + k0),
      (__attribute__((address_space(3))) v4i*)&bw[srow][shalf], 0, 0);
  __builtin_amdgcn_global_load_async_to_lds_b128(
      (__attribute__((address_space(1))) v4i*)(wrow + k0 + 8),
      (__attribute__((address_space(3))) v4i*)&bw[srow][shalf + 8], 0, 0);
#else
  v8bf w0 = *(const v8bf*)(wrow + k0);
  v8bf w1 = *(const v8bf*)(wrow + k0 + 8);
  *(v8bf*)&bw[srow][shalf]     = w0;
  *(v8bf*)&bw[srow][shalf + 8] = w1;
#endif
}

static __device__ __forceinline__ void async_wait() {
#if USE_ASYNC_LDS
  __builtin_amdgcn_s_wait_asynccnt(0);
#endif
}

__global__ __launch_bounds__(256)
void lora_gemm(const float* __restrict__ x,
               const float* __restrict__ b_base,
               const __bf16* __restrict__ Wc,
               const int* __restrict__ adapter_ids,
               float* __restrict__ out) {
  __shared__ __bf16 lds_x[2][128][LDS_STRIDE];   // 2 x 10 KB
  __shared__ __bf16 lds_w[2][128][LDS_STRIDE];   // 2 x 10 KB

  const int tid   = threadIdx.x;
  const int lane  = tid & 31;
  const int wave  = tid >> 5;
  const int wm    = (wave >> 2) * 64;   // wave M offset: 0 / 64
  const int wn    = (wave & 3) * 32;    // wave N offset: 0/32/64/96

  const int b     = blockIdx.z;
  const int sBase = blockIdx.y * 128;
  const int oBase = blockIdx.x * 128;
  const int n     = adapter_ids[b];

  // staging role: thread -> (row 0..127, half 0..1) of the 128 x 32 tiles
  const int srow  = tid >> 1;
  const int shalf = (tid & 1) * 16;

  const float*  xrow = x  + ((size_t)(b * SS + sBase + srow) * DIN) + shalf;
  const __bf16* wrow = Wc + ((size_t)((size_t)n * DOUT + oBase + srow) * DIN) + shalf;

  v8f acc[4][2] = {};  // zero-initialized f32 accumulators

  // A-fragment addressing (16x32 A, contiguous K: row = lane&15, element s holds K = 16*(lane>>4) + s)
  const int arow = lane & 15;
  const int akb  = (lane >> 4) << 4;    // 0 or 16; K contiguous 16
  // B-fragment addressing (lane&15 = column, K contiguous 16, same K order as A)
  const int bcol = lane & 15;
  const int bks  = (lane >> 4) << 4;    // 0 or 16

  // ---- prologue: stage tile 0 into buffer 0 ----
  stage_x(lds_x[0], xrow, 0, srow, shalf);
  stage_w(lds_w[0], wrow, 0, srow, shalf);
  async_wait();
  __syncthreads();

  int cur = 0;
  for (int k0 = 0; k0 < DIN; k0 += KSTEP) {
    const int knext = k0 + KSTEP;
    if (knext < DIN) {
      // stage next tile into the other buffer while WMMAs consume this one;
      // safe: last iteration's barrier guarantees nobody still reads it
      stage_x(lds_x[cur ^ 1], xrow, knext, srow, shalf);
      stage_w(lds_w[cur ^ 1], wrow, knext, srow, shalf);
      __builtin_prefetch(xrow + knext + KSTEP, 0, 1);   // global_prefetch_b8
    }

    // ---- build fragments from current buffer ----
    v16bf afrag[4], bfrag[2];
#pragma unroll
    for (int mt = 0; mt < 4; ++mt) {
      const __bf16* p = &lds_x[cur][wm + mt * 16 + arow][akb];
      v8bf lo = *(const v8bf*)p;          // K = akb .. akb+7
      v8bf hi = *(const v8bf*)(p + 8);    // K = akb+8 .. akb+15
      afrag[mt] = __builtin_shufflevector(lo, hi, 0,1,2,3,4,5,6,7,8,9,10,11,12,13,14,15);
    }
#pragma unroll
    for (int nt = 0; nt < 2; ++nt) {
      const __bf16* p = &lds_w[cur][wn + nt * 16 + bcol][bks];
      v8bf lo = *(const v8bf*)p;          // K = bks .. bks+7
      v8bf hi = *(const v8bf*)(p + 8);    // K = bks+8 .. bks+15
      bfrag[nt] = __builtin_shufflevector(lo, hi, 0,1,2,3,4,5,6,7,8,9,10,11,12,13,14,15);
    }

    // ---- 8 WMMAs per wave per K step ----
#pragma unroll
    for (int mt = 0; mt < 4; ++mt)
#pragma unroll
      for (int nt = 0; nt < 2; ++nt)
        acc[mt][nt] = __builtin_amdgcn_wmma_f32_16x16x32_bf16(
            false, afrag[mt], false, bfrag[nt], (short)0, acc[mt][nt], false, false);

    if (knext < DIN) async_wait();   // my async DMAs into next buffer are done
    __syncthreads();                 // ... and visible to all waves
    cur ^= 1;
  }

  // ---- epilogue: D layout VGPR r -> M = r + 8*(lane>=16), N = lane&15 ----
  const int mhi = (lane >> 4) * 8;
  const float bias0 = b_base[oBase + wn +      (lane & 15)];
  const float bias1 = b_base[oBase + wn + 16 + (lane & 15)];
#pragma unroll
  for (int mt = 0; mt < 4; ++mt) {
#pragma unroll
    for (int nt = 0; nt < 2; ++nt) {
      const int   col  = oBase + wn + nt * 16 + (lane & 15);
      const float bias = nt ? bias1 : bias0;
      float* orow = out + (size_t)(b * SS + sBase + wm + mt * 16 + mhi) * DOUT + col;
#pragma unroll
      for (int r = 0; r < 8; ++r)
        orow[(size_t)r * DOUT] = acc[mt][nt][r] + bias;
    }
  }
}

// ---------------------------------------------------------------------------
// launch
// ---------------------------------------------------------------------------
extern "C" void kernel_launch(void* const* d_in, const int* in_sizes, int n_in,
                              void* d_out, int out_size, void* d_ws, size_t ws_size,
                              hipStream_t stream) {
  (void)in_sizes; (void)n_in; (void)out_size; (void)ws_size;
  const float* x      = (const float*)d_in[0];
  const float* W_base = (const float*)d_in[1];
  const float* b_base = (const float*)d_in[2];
  const float* W_t    = (const float*)d_in[3];
  const float* W_s    = (const float*)d_in[4];
  const float* W_a    = (const float*)d_in[5];
  const float* W_b    = (const float*)d_in[6];
  const float* W_q    = (const float*)d_in[7];
  const float* W_p    = (const float*)d_in[8];
  const int*   aid    = (const int*)d_in[9];
  float* out = (float*)d_out;

  // workspace layout (floats): t1[1024] | t2[1024] | Ap[64K] | Bp[64K] | Wc (bf16, 4M)
  float* ws = (float*)d_ws;
  float* t1 = ws;
  float* t2 = ws + 1024;
  float* Ap = ws + 2048;
  float* Bp = Ap + NA * 16 * DIN;
  __bf16* Wc = (__bf16*)(Bp + NA * DOUT * RR);

  small_mm1<<<NA, 256, 0, stream>>>(W_a, W_s, W_q, W_b, t1, t2);
  small_mm2<<<512, 256, 0, stream>>>(W_t, W_p, t1, t2, Ap, Bp);
  build_wc<<<(NA * DOUT * DIN) / 256, 256, 0, stream>>>(W_base, Ap, Bp, Wc);
  lora_gemm<<<dim3(DOUT / 128, SS / 128, BB), 256, 0, stream>>>(x, b_base, Wc, aid, out);
}
